// NanoLLM_52123723104798
// MI455X (gfx1250) — matrix operand order repaired and checked by reading, not compiled, not run on hardware
//
#include <hip/hip_runtime.h>
#include <math.h>

// ---------------- model constants ----------------
#define B_   2
#define S_   2048
#define V_   32000
#define D_   1024
#define H_   16
#define DH_  64
#define DFF_ 4096
#define L_   8
#define M_   (B_*S_)   // 4096 token rows

typedef __attribute__((ext_vector_type(16))) _Float16 v16h;
typedef __attribute__((ext_vector_type(8)))  _Float16 v8h;
typedef __attribute__((ext_vector_type(8)))  float    v8f;
typedef __attribute__((ext_vector_type(4)))  unsigned int u32x4;
typedef __attribute__((ext_vector_type(8)))  int      i32x8;
typedef __attribute__((ext_vector_type(4)))  int      i32x4;

__device__ __forceinline__ v16h nano_cat8(v8h lo, v8h hi) {
  return __builtin_shufflevector(lo, hi, 0,1,2,3,4,5,6,7,8,9,10,11,12,13,14,15);
}
__device__ __forceinline__ v8f nano_zero8() {
  v8f z;
#pragma unroll
  for (int i = 0; i < 8; ++i) z[i] = 0.0f;
  return z;
}

// ---- CDNA5 async global->LDS copy (ASYNCcnt-tracked, no VGPR staging) ----
__device__ __forceinline__ void nano_async_b128(unsigned int lds,
                                                const void* g) {
  asm volatile("global_load_async_to_lds_b128 %0, %1, off"
               :: "v"(lds), "v"((unsigned long long)(uintptr_t)g) : "memory");
}
__device__ __forceinline__ void nano_async_b128_o16(unsigned int lds,
                                                    const void* g) {
  asm volatile("global_load_async_to_lds_b128 %0, %1, off offset:16"
               :: "v"(lds), "v"((unsigned long long)(uintptr_t)g) : "memory");
}
__device__ __forceinline__ void nano_wait_async() {
  asm volatile("s_wait_asynccnt 0x0" ::: "memory");
}

// ---- CDNA5 Tensor Data Mover: 2D tile (rows x 32 halves) global -> padded LDS ----
// LDS row pitch 80B: pad_interval=3 (pad after 16 DWORDs = 64B), pad_amount=3 (4 DWORDs = 16B).
// Toolchain here is the 6-arg clang-23 form:
//   (u32x4 g0, i32x8 g1, i32x4 g2, i32x4 g3, i32x8 g4, i32 cpol)
__device__ __forceinline__ void nano_tdm_load_tile(unsigned int lds_addr,
                                                   const void* gaddr,
                                                   unsigned int k_elems,      // tensor_dim0 / stride
                                                   unsigned int rows,         // tensor_dim1 == tile_dim1
                                                   unsigned int tile_k) {     // tile_dim0 (elements)
  unsigned long long ga = (unsigned long long)(uintptr_t)gaddr;
  u32x4 g0;
  g0[0] = 1u;                                            // count=1, user descriptor
  g0[1] = lds_addr;                                      // lds_addr [63:32]
  g0[2] = (unsigned int)(ga & 0xFFFFFFFFu);              // global_addr lo
  g0[3] = (unsigned int)((ga >> 32) & 0x01FFFFFFu) | 0x80000000u;  // addr hi | type=2
  i32x8 g1;
  g1[0] = (int)((1u << 16) | (1u << 20) | (3u << 22) | (3u << 25)); // data_size=2B, pad on
  g1[1] = (int)((k_elems & 0xFFFFu) << 16);              // tensor_dim0[15:0]
  g1[2] = (int)((k_elems >> 16) | ((rows & 0xFFFFu) << 16));  // dim0 hi | tensor_dim1 lo
  g1[3] = (int)((rows >> 16) | (tile_k << 16));          // dim1 hi | tile_dim0
  g1[4] = (int)rows;                                     // tile_dim1 (tile_dim2=0)
  g1[5] = (int)k_elems;                                  // tensor_dim0_stride lo
  g1[6] = 0;                                             // stride hi | dim1_stride lo
  g1[7] = 0;
  i32x4 z4;
  z4[0] = 0; z4[1] = 0; z4[2] = 0; z4[3] = 0;
  i32x8 z8;
#pragma unroll
  for (int i = 0; i < 8; ++i) z8[i] = 0;
  __builtin_amdgcn_tensor_load_to_lds(g0, g1, z4, z4, z8, 0);
}
__device__ __forceinline__ void nano_wait_tensor() {
  __builtin_amdgcn_s_wait_tensorcnt((short)0);
}

// ---------------- block reductions (256 threads) ----------------
__device__ __forceinline__ float nano_blk_sum(float v, float* red) {
  int t = threadIdx.x;
  red[t] = v; __syncthreads();
  for (int s = 128; s > 0; s >>= 1) { if (t < s) red[t] += red[t + s]; __syncthreads(); }
  float r = red[0]; __syncthreads();
  return r;
}
__device__ __forceinline__ float nano_blk_max(float v, float* red) {
  int t = threadIdx.x;
  red[t] = v; __syncthreads();
  for (int s = 128; s > 0; s >>= 1) { if (t < s) red[t] = fmaxf(red[t], red[t + s]); __syncthreads(); }
  float r = red[0]; __syncthreads();
  return r;
}

// ---------------- embedding: x = tok_emb[id] + pos_emb[s] ----------------
__global__ void __launch_bounds__(256) nano_embed(const int* __restrict__ ids,
                                                  const float* __restrict__ tok,
                                                  const float* __restrict__ pos,
                                                  float* __restrict__ x) {
  int row = blockIdx.x;              // 0..M_-1
  int s   = row & (S_ - 1);
  int id  = ids[row];
  const float* tp = tok + (size_t)id * D_;
  const float* pp = pos + (size_t)s  * D_;
  float* xp = x + (size_t)row * D_;
#pragma unroll
  for (int i = 0; i < 4; ++i) {
    int c = threadIdx.x + i * 256;
    xp[c] = tp[c] + pp[c];
  }
}

// ---------------- fp32 -> fp16 cast (tok_emb for logits GEMM) ----------------
__global__ void __launch_bounds__(256) nano_castf16(const float* __restrict__ in,
                                                    _Float16* __restrict__ out, size_t n) {
  for (size_t i = (size_t)blockIdx.x * blockDim.x + threadIdx.x; i < n;
       i += (size_t)gridDim.x * blockDim.x)
    out[i] = (_Float16)in[i];
}

// ---------------- weight quantization (BitNet ternary) ----------------
__global__ void __launch_bounds__(256) nano_absmean_partial(const float* __restrict__ w,
                                                            size_t n, float* __restrict__ part) {
  __shared__ float red[256];
  float s = 0.0f;
  for (size_t i = (size_t)blockIdx.x * blockDim.x + threadIdx.x; i < n;
       i += (size_t)gridDim.x * blockDim.x)
    s += fabsf(w[i]);
  float tot = nano_blk_sum(s, red);
  if (threadIdx.x == 0) part[blockIdx.x] = tot;
}
__global__ void __launch_bounds__(256) nano_absmean_final(const float* __restrict__ part,
                                                          size_t n, float* __restrict__ scale) {
  __shared__ float red[256];
  float tot = nano_blk_sum(part[threadIdx.x], red);
  if (threadIdx.x == 0) {
    float mean = tot / (float)n;
    scale[0] = 1.0f / fmaxf(mean, 1e-5f);
  }
}
__global__ void __launch_bounds__(256) nano_wquant(const float* __restrict__ w,
                                                   const float* __restrict__ scale,
                                                   _Float16* __restrict__ out, size_t n) {
  float sc = scale[0];
  float inv = 1.0f / sc;
  for (size_t i = (size_t)blockIdx.x * blockDim.x + threadIdx.x; i < n;
       i += (size_t)gridDim.x * blockDim.x) {
    float q = fminf(fmaxf(rintf(w[i] * sc), -1.0f), 1.0f) * inv;
    out[i] = (_Float16)q;
  }
}

// ---------------- LayerNorm (+ optional act_quant) -> fp16 ----------------
// MODE 0: LN + act_quant ; MODE 1: LN only (final LN before logits)
template <int MODE>
__global__ void __launch_bounds__(256) nano_ln_quant(const float* __restrict__ x,
                                                     const float* __restrict__ w,
                                                     _Float16* __restrict__ out) {
  __shared__ float red[256];
  const float* row = x + (size_t)blockIdx.x * D_;
  _Float16* orow = out + (size_t)blockIdx.x * D_;
  int t = threadIdx.x;
  float loc[4];
#pragma unroll
  for (int i = 0; i < 4; ++i) loc[i] = row[t + i * 256];
  float mu = nano_blk_sum(loc[0] + loc[1] + loc[2] + loc[3], red) * (1.0f / D_);
  float vs = 0.0f;
#pragma unroll
  for (int i = 0; i < 4; ++i) { float d = loc[i] - mu; vs += d * d; }
  float var = nano_blk_sum(vs, red) * (1.0f / D_);
  float rs = rsqrtf(var + 1e-5f);
  float y[4];
#pragma unroll
  for (int i = 0; i < 4; ++i) y[i] = (loc[i] - mu) * rs * w[t + i * 256];
  if (MODE == 0) {
    float am = 0.0f;
#pragma unroll
    for (int i = 0; i < 4; ++i) am = fmaxf(am, fabsf(y[i]));
    float amax = nano_blk_max(am, red);
    float sc = 127.0f / fmaxf(amax, 1e-5f);
    float inv = 1.0f / sc;
#pragma unroll
    for (int i = 0; i < 4; ++i) {
      float q = fminf(fmaxf(rintf(y[i] * sc), -128.0f), 127.0f) * inv;
      orow[t + i * 256] = (_Float16)q;
    }
  } else {
#pragma unroll
    for (int i = 0; i < 4; ++i) orow[t + i * 256] = (_Float16)y[i];
  }
}

// ---------------- row act_quant (MODE 1: relu^2 first) -> fp16 ----------------
template <int RL, int MODE>
__global__ void __launch_bounds__(256) nano_rowquant(const float* __restrict__ in,
                                                     _Float16* __restrict__ out) {
  __shared__ float red[256];
  const int PER = RL / 256;
  const float* row = in + (size_t)blockIdx.x * RL;
  _Float16* orow = out + (size_t)blockIdx.x * RL;
  int t = threadIdx.x;
  float loc[PER];
  float am = 0.0f;
#pragma unroll
  for (int j = 0; j < PER; ++j) {
    float v = row[t + j * 256];
    if (MODE == 1) { v = fmaxf(v, 0.0f); v = v * v; }
    loc[j] = v;
    am = fmaxf(am, fabsf(v));
  }
  float amax = nano_blk_max(am, red);
  float sc = 127.0f / fmaxf(amax, 1e-5f);
  float inv = 1.0f / sc;
#pragma unroll
  for (int j = 0; j < PER; ++j) {
    float q = fminf(fmaxf(rintf(loc[j] * sc), -128.0f), 127.0f) * inv;
    orow[t + j * 256] = (_Float16)q;
  }
}

// ---------------- RoPE: qkv(fp32, B,S,3*D) -> Qh/Kh/Vh (fp16, B,H,S,DH) ----------------
__global__ void __launch_bounds__(256) nano_rope(const float* __restrict__ qkv,
                                                 _Float16* __restrict__ Qh,
                                                 _Float16* __restrict__ Kh,
                                                 _Float16* __restrict__ Vh) {
  int row = blockIdx.x;            // b*S + s
  int b = row >> 11;               // /S_
  int s = row & (S_ - 1);
  const float* r = qkv + (size_t)row * (3 * D_);
  int t = threadIdx.x;
  for (int p = t; p < H_ * (DH_ / 2); p += 256) {   // 512 (h,d) pairs
    int h = p >> 5, d = p & 31;
    float invf = __expf(-9.210340371976184f * (float)d * (1.0f / 32.0f)); // 10000^(-d/32)
    float ang = (float)s * invf;
    float cs = __cosf(ang), sn = __sinf(ang);
    size_t ob = (((size_t)(b * H_ + h)) * S_ + s) * DH_;
    float q1 = r[h * DH_ + d], q2 = r[h * DH_ + d + 32];
    Qh[ob + d]      = (_Float16)((q1 * cs - q2 * sn) * 0.125f);  // fold 1/sqrt(DH)
    Qh[ob + d + 32] = (_Float16)((q1 * sn + q2 * cs) * 0.125f);
    float k1 = r[D_ + h * DH_ + d], k2 = r[D_ + h * DH_ + d + 32];
    Kh[ob + d]      = (_Float16)(k1 * cs - k2 * sn);
    Kh[ob + d + 32] = (_Float16)(k1 * sn + k2 * cs);
  }
  for (int i = t; i < D_; i += 256) {
    int h = i >> 6, d = i & 63;
    Vh[(((size_t)(b * H_ + h)) * S_ + s) * DH_ + d] = (_Float16)r[2 * D_ + i];
  }
}

// ---------------- WMMA GEMM: C[M,N] = A[M,K](f16) * W[N,K]^T(f16) (+res) ----------------
// block tile 128x128, 8 waves (wave32), each wave 32x64 = 2x4 fragments of 16x16.
// A tile: per-lane async-to-LDS (ASYNCcnt). W tile: TDM tensor_load_to_lds (TENSORcnt),
// issued by wave 0 with pad-to-80B rows. Double-buffered; DMA overlaps WMMA.
__global__ void __launch_bounds__(256) nano_gemm_wmma(const _Float16* __restrict__ A,
                                                      const _Float16* __restrict__ W,
                                                      float* __restrict__ C,
                                                      const float* __restrict__ res,
                                                      int N, int K) {
  __shared__ __align__(16) _Float16 As[2][128 * 40];
  __shared__ __align__(16) _Float16 Ws[2][128 * 40];
  int t = threadIdx.x;
  int lane = t & 31, wid = t >> 5;
  int hi = lane >> 4, ln = lane & 15;
  int m0g = blockIdx.y * 128, n0g = blockIdx.x * 128;
  int wm = (wid >> 1) * 32;   // 0,32,64,96
  int wn = (wid & 1) * 64;    // 0,64
  int lrow = t >> 1, lseg = (t & 1) * 16;

  v8f acc[2][4];
#pragma unroll
  for (int i = 0; i < 2; ++i)
#pragma unroll
    for (int j = 0; j < 4; ++j) acc[i][j] = nano_zero8();

  const _Float16* Ag = A + (size_t)(m0g + lrow) * K + lseg;
  const _Float16* Wt = W + (size_t)n0g * K;         // block's weight-tile rows (uniform)
  unsigned int ldsA[2], ldsWb[2];
#pragma unroll
  for (int bfi = 0; bfi < 2; ++bfi) {
    ldsA[bfi]  = (unsigned int)(uintptr_t)(&As[bfi][lrow * 40 + lseg]);
    ldsWb[bfi] = (unsigned int)(uintptr_t)(&Ws[bfi][0]);
  }

  // preload tile 0
  nano_async_b128(ldsA[0], Ag);
  nano_async_b128_o16(ldsA[0], Ag);
  if (wid == 0) {
    nano_tdm_load_tile(ldsWb[0], Wt, (unsigned)K, 128u, 32u);
  }
  nano_wait_async();
  if (wid == 0) nano_wait_tensor();
  __syncthreads();

  for (int k0 = 0; k0 < K; k0 += 32) {
    int cur = (k0 >> 5) & 1, nxt = cur ^ 1;
    if (k0 + 32 < K) {          // kick off DMA for next tile (overlaps WMMA below)
      nano_async_b128(ldsA[nxt], Ag + k0 + 32);
      nano_async_b128_o16(ldsA[nxt], Ag + k0 + 32);
      if (wid == 0)
        nano_tdm_load_tile(ldsWb[nxt], Wt + k0 + 32, (unsigned)K, 128u, 32u);
    }
    const _Float16* Asc = &As[cur][0];
    const _Float16* Wsc = &Ws[cur][0];
    v16h af[2], bf[4];
#pragma unroll
    for (int fm = 0; fm < 2; ++fm) {       // A 16x32 fragment layout
      const _Float16* rp = Asc + (wm + fm * 16 + ln) * 40;
      v8h lo = *(const v8h*)(rp + hi * 8);
      v8h hh = *(const v8h*)(rp + 16 + hi * 8);
      af[fm] = nano_cat8(lo, hh);
    }
#pragma unroll
    for (int fn = 0; fn < 4; ++fn) {       // B 32x16 fragment layout (W row = B column)
      const _Float16* rp = Wsc + (wn + fn * 16 + ln) * 40 + hi * 8;
      bf[fn] = nano_cat8(*(const v8h*)rp, *(const v8h*)(rp + 16));
    }
#pragma unroll
    for (int fm = 0; fm < 2; ++fm)
#pragma unroll
      for (int fn = 0; fn < 4; ++fn)
        acc[fm][fn] = __builtin_amdgcn_wmma_f32_16x16x32_f16(
            false, af[fm], false, bf[fn], (short)0, acc[fm][fn], false, false);
    nano_wait_async();
    if (wid == 0) nano_wait_tensor();
    __syncthreads();
  }

#pragma unroll
  for (int fm = 0; fm < 2; ++fm)
#pragma unroll
    for (int fn = 0; fn < 4; ++fn) {
      int n = n0g + wn + fn * 16 + ln;
#pragma unroll
      for (int v = 0; v < 8; ++v) {
        int m = m0g + wm + fm * 16 + hi * 8 + v;
        size_t idx = (size_t)m * N + n;
        float val = acc[fm][fn][v];
        if (res) val += res[idx];
        C[idx] = val;
      }
    }
}

// ---------------- flash attention (WMMA QK^T and P*V, online softmax) ----------------
// grid (S/128, H, B), 8 waves/block, each wave owns 16 query rows.
// K/V tiles (32x64) via async-to-LDS, double-buffered.
__global__ void __launch_bounds__(256) nano_flash_attn(const _Float16* __restrict__ Qh,
                                                       const _Float16* __restrict__ Kh,
                                                       const _Float16* __restrict__ Vh,
                                                       float* __restrict__ O) {
  __shared__ __align__(16) _Float16 Ks[2][32 * 64];
  __shared__ __align__(16) _Float16 Vs[2][32 * 64];
  __shared__ __align__(16) _Float16 Ps[8 * 16 * 40];
  int t = threadIdx.x, lane = t & 31, wid = t >> 5;
  int hi = lane >> 4, ln = lane & 15;
  int h = blockIdx.y, b = blockIdx.z;
  int q0b = blockIdx.x * 128;
  int q0 = q0b + wid * 16;
  const _Float16* Qp = Qh + ((size_t)(b * H_ + h)) * S_ * DH_;
  const _Float16* Kp = Kh + ((size_t)(b * H_ + h)) * S_ * DH_;
  const _Float16* Vp = Vh + ((size_t)(b * H_ + h)) * S_ * DH_;

  // Q fragments: A-matrix 16x64 -> two 16x32 frags
  v16h qa[2];
#pragma unroll
  for (int kc = 0; kc < 2; ++kc) {
    const _Float16* rp = Qp + (size_t)(q0 + ln) * DH_ + kc * 32;
    v8h lo = *(const v8h*)(rp + hi * 8);
    v8h hh = *(const v8h*)(rp + 16 + hi * 8);
    qa[kc] = nano_cat8(lo, hh);
  }

  v8f oacc[4];
#pragma unroll
  for (int i = 0; i < 4; ++i) oacc[i] = nano_zero8();
  float mstat[8], lstat[8];
#pragma unroll
  for (int v = 0; v < 8; ++v) { mstat[v] = -3.0e38f; lstat[v] = 0.0f; }

  int ktiles = (q0b + 128) / 32;
  int krow = t >> 3, kc8 = (t & 7) * 8;   // 16B chunk per thread per matrix
  _Float16* pp = Ps + wid * 16 * 40;
  unsigned int ldsK[2], ldsV[2];
#pragma unroll
  for (int bfi = 0; bfi < 2; ++bfi) {
    ldsK[bfi] = (unsigned int)(uintptr_t)(&Ks[bfi][krow * 64 + kc8]);
    ldsV[bfi] = (unsigned int)(uintptr_t)(&Vs[bfi][krow * 64 + kc8]);
  }

  // preload key-tile 0
  nano_async_b128(ldsK[0], Kp + (size_t)krow * DH_ + kc8);
  nano_async_b128(ldsV[0], Vp + (size_t)krow * DH_ + kc8);
  nano_wait_async();
  __syncthreads();

  for (int kt = 0; kt < ktiles; ++kt) {
    int cur = kt & 1, nxt = cur ^ 1;
    if (kt + 1 < ktiles) {      // DMA next K/V tile while computing this one
      nano_async_b128(ldsK[nxt], Kp + (size_t)((kt + 1) * 32 + krow) * DH_ + kc8);
      nano_async_b128(ldsV[nxt], Vp + (size_t)((kt + 1) * 32 + krow) * DH_ + kc8);
    }
    const _Float16* Ksc = &Ks[cur][0];
    const _Float16* Vsc = &Vs[cur][0];

    // scores: S(16x32) = Q(16x64) * K^T(64x32)
    v8f sc[2];
    sc[0] = nano_zero8(); sc[1] = nano_zero8();
#pragma unroll
    for (int nf = 0; nf < 2; ++nf)
#pragma unroll
      for (int kc = 0; kc < 2; ++kc) {
        const _Float16* rp = Ksc + (size_t)(nf * 16 + ln) * 64 + kc * 32 + hi * 8;
        v16h bk = nano_cat8(*(const v8h*)rp, *(const v8h*)(rp + 16));
        sc[nf] = __builtin_amdgcn_wmma_f32_16x16x32_f16(
            false, qa[kc], false, bk, (short)0, sc[nf], false, false);
      }

    // causal mask + online softmax
    float mt[8];
#pragma unroll
    for (int v = 0; v < 8; ++v) {
      int q = q0 + hi * 8 + v;
      float s0 = ((kt * 32 + ln)      <= q) ? sc[0][v] : -1.0e9f;
      float s1 = ((kt * 32 + 16 + ln) <= q) ? sc[1][v] : -1.0e9f;
      sc[0][v] = s0; sc[1][v] = s1;
      mt[v] = fmaxf(s0, s1);
    }
#pragma unroll
    for (int v = 0; v < 8; ++v)
      for (int mk = 1; mk < 16; mk <<= 1)
        mt[v] = fmaxf(mt[v], __shfl_xor(mt[v], mk, 32));
    float alpha[8];
#pragma unroll
    for (int v = 0; v < 8; ++v) {
      float mnew = fmaxf(mstat[v], mt[v]);
      alpha[v] = __expf(mstat[v] - mnew);
      mstat[v] = mnew;
    }
    float rowsum[8];
#pragma unroll
    for (int v = 0; v < 8; ++v) {
      float p0 = __expf(sc[0][v] - mstat[v]);
      float p1 = __expf(sc[1][v] - mstat[v]);
      rowsum[v] = p0 + p1;
      int m = hi * 8 + v;
      pp[m * 40 + ln]      = (_Float16)p0;   // restage P into A-frag layout
      pp[m * 40 + 16 + ln] = (_Float16)p1;
    }
#pragma unroll
    for (int v = 0; v < 8; ++v) {
      for (int mk = 1; mk < 16; mk <<= 1)
        rowsum[v] += __shfl_xor(rowsum[v], mk, 32);
      lstat[v] = lstat[v] * alpha[v] + rowsum[v];
    }
#pragma unroll
    for (int of = 0; of < 4; ++of)
#pragma unroll
      for (int v = 0; v < 8; ++v) oacc[of][v] *= alpha[v];

    // O(16x64) += P(16x32) * V(32x64)  (same-wave LDS RAW: DS ops are in-order)
    v16h pa;
    {
      const _Float16* rp = pp + ln * 40;
      v8h lo = *(const v8h*)(rp + hi * 8);
      v8h hh = *(const v8h*)(rp + 16 + hi * 8);
      pa = nano_cat8(lo, hh);
    }
#pragma unroll
    for (int of = 0; of < 4; ++of) {
      v16h bv;
#pragma unroll
      for (int j = 0; j < 16; ++j)
        bv[j] = Vsc[(hi * 8 + j + (j >= 8 ? 8 : 0)) * 64 + of * 16 + ln];
      oacc[of] = __builtin_amdgcn_wmma_f32_16x16x32_f16(
          false, pa, false, bv, (short)0, oacc[of], false, false);
    }
    nano_wait_async();
    __syncthreads();
  }

  // normalize + write (B,S,H*DH)
#pragma unroll
  for (int of = 0; of < 4; ++of)
#pragma unroll
    for (int v = 0; v < 8; ++v) {
      int q = q0 + hi * 8 + v;
      O[((size_t)(b * S_ + q)) * D_ + h * DH_ + of * 16 + ln] = oacc[of][v] / lstat[v];
    }
}

// ---------------- host-side orchestration ----------------
extern "C" void kernel_launch(void* const* d_in, const int* in_sizes, int n_in,
                              void* d_out, int out_size, void* d_ws, size_t ws_size,
                              hipStream_t stream) {
  (void)in_sizes; (void)n_in; (void)out_size; (void)ws_size;
  const int*   ids    = (const int*)d_in[0];
  const float* tokemb = (const float*)d_in[1];
  const float* posemb = (const float*)d_in[2];
  const float* ln1w   = (const float*)d_in[3];
  const float* ln2w   = (const float*)d_in[4];
  const float* Wqkv   = (const float*)d_in[5];
  const float* Wo     = (const float*)d_in[6];
  const float* W1     = (const float*)d_in[7];
  const float* W2     = (const float*)d_in[8];
  const float* lnfw   = (const float*)d_in[9];

  char* p = (char*)d_ws;
  auto carve = [&](size_t bytes) -> void* {
    void* r = (void*)p;
    p += (bytes + 255) & ~(size_t)255;
    return r;
  };
  float*    x      = (float*)carve((size_t)M_ * D_ * 4);
  _Float16* xq     = (_Float16*)carve((size_t)M_ * D_ * 2);
  float*    qkv    = (float*)carve((size_t)M_ * 3 * D_ * 4);
  _Float16* Qh     = (_Float16*)carve((size_t)B_ * H_ * S_ * DH_ * 2);
  _Float16* Kh     = (_Float16*)carve((size_t)B_ * H_ * S_ * DH_ * 2);
  _Float16* Vh     = (_Float16*)carve((size_t)B_ * H_ * S_ * DH_ * 2);
  float*    attn   = (float*)carve((size_t)M_ * D_ * 4);
  float*    hbuf   = (float*)carve((size_t)M_ * DFF_ * 4);
  _Float16* hq     = (_Float16*)carve((size_t)M_ * DFF_ * 2);
  _Float16* wqkv_q = (_Float16*)carve((size_t)3 * D_ * D_ * 2);
  _Float16* wo_q   = (_Float16*)carve((size_t)D_ * D_ * 2);
  _Float16* w1_q   = (_Float16*)carve((size_t)DFF_ * D_ * 2);
  _Float16* w2_q   = (_Float16*)carve((size_t)D_ * DFF_ * 2);
  _Float16* tokh   = (_Float16*)carve((size_t)V_ * D_ * 2);
  float*    parts  = (float*)carve(256 * 4);
  float*    scalef = (float*)carve(4);

  dim3 blk(256);
  nano_embed<<<M_, blk, 0, stream>>>(ids, tokemb, posemb, x);
  nano_castf16<<<4096, blk, 0, stream>>>(tokemb, tokh, (size_t)V_ * D_);

  auto qweight = [&](const float* w, size_t n, _Float16* out) {
    nano_absmean_partial<<<256, blk, 0, stream>>>(w, n, parts);
    nano_absmean_final<<<1, blk, 0, stream>>>(parts, n, scalef);
    nano_wquant<<<2048, blk, 0, stream>>>(w, scalef, out, n);
  };

  for (int l = 0; l < L_; ++l) {
    qweight(Wqkv + (size_t)l * 3 * D_ * D_, (size_t)3 * D_ * D_, wqkv_q);
    qweight(Wo   + (size_t)l * D_ * D_,     (size_t)D_ * D_,     wo_q);
    qweight(W1   + (size_t)l * DFF_ * D_,   (size_t)DFF_ * D_,   w1_q);
    qweight(W2   + (size_t)l * D_ * DFF_,   (size_t)D_ * DFF_,   w2_q);

    // attention
    nano_ln_quant<0><<<M_, blk, 0, stream>>>(x, ln1w + l * D_, xq);
    nano_gemm_wmma<<<dim3(3 * D_ / 128, M_ / 128), blk, 0, stream>>>(
        xq, wqkv_q, qkv, nullptr, 3 * D_, D_);
    nano_rope<<<M_, blk, 0, stream>>>(qkv, Qh, Kh, Vh);
    nano_flash_attn<<<dim3(S_ / 128, H_, B_), blk, 0, stream>>>(Qh, Kh, Vh, attn);
    nano_rowquant<D_, 0><<<M_, blk, 0, stream>>>(attn, xq);
    nano_gemm_wmma<<<dim3(D_ / 128, M_ / 128), blk, 0, stream>>>(
        xq, wo_q, x, x, D_, D_);

    // ffn
    nano_ln_quant<0><<<M_, blk, 0, stream>>>(x, ln2w + l * D_, xq);
    nano_gemm_wmma<<<dim3(DFF_ / 128, M_ / 128), blk, 0, stream>>>(
        xq, w1_q, hbuf, nullptr, DFF_, D_);
    nano_rowquant<DFF_, 1><<<M_, blk, 0, stream>>>(hbuf, hq);
    nano_gemm_wmma<<<dim3(D_ / 128, M_ / 128), blk, 0, stream>>>(
        hq, w2_q, x, x, D_, DFF_);
  }

  // final LN + logits
  nano_ln_quant<1><<<M_, blk, 0, stream>>>(x, lnfw, xq);
  nano_gemm_wmma<<<dim3(V_ / 128, M_ / 128), blk, 0, stream>>>(
      xq, tokh, (float*)d_out, nullptr, V_, D_);
}
